// MultiHeadAttention_35235911696407
// MI455X (gfx1250) — matrix operand order repaired and checked by reading, not compiled, not run on hardware
//
#include <hip/hip_runtime.h>
#include <hip/hip_bf16.h>

// ---------------- CDNA5 WMMA types ----------------
typedef __attribute__((ext_vector_type(16))) __bf16 v16bf;
typedef __attribute__((ext_vector_type(8)))  float  v8f;

union FragAB { v16bf v; unsigned int d[8]; unsigned short u[16]; };
union FragC  { v8f   v; float f[8]; };
union Pack8  { uint4 q; unsigned int d[4]; unsigned short u[8]; };

#define BATCH 4
#define SEQ   2048
#define EMB   1024
#define NH    16
#define HD    64
#define HDIM  1024   // NH*HD
#define LPAD  40     // LDS row stride in shorts (80 B, 16B-aligned rows)

__device__ __forceinline__ unsigned short f2bf(float f) {
  unsigned int u = __float_as_uint(f);
  u += 0x7FFFu + ((u >> 16) & 1u);          // round-to-nearest-even
  return (unsigned short)(u >> 16);
}

__device__ __forceinline__ v8f wmma_bf16(const FragAB& a, const FragAB& b, v8f c) {
  return __builtin_amdgcn_wmma_f32_16x16x32_bf16(
      false, a.v, false, b.v, (short)0, c, false, false);
}

// A-fragment (16-bit A 16x32, ISA 7.12.2): per lane the 8 dwords live at
// byte offsets {0..15} and {32..47}, +16*half  -> two b128 loads.
__device__ __forceinline__ void load_fragA(FragAB& a, const unsigned short* rowp, int hf) {
  const uint4* p = (const uint4*)(rowp + 8 * hf);
  uint4 lo = p[0], hi = p[2];
  a.d[0] = lo.x; a.d[1] = lo.y; a.d[2] = lo.z; a.d[3] = lo.w;
  a.d[4] = hi.x; a.d[5] = hi.y; a.d[6] = hi.z; a.d[7] = hi.w;
}
// B-fragment (same K order as A): bytes {0..15} and {32..47}, +16*half -> two b128 loads.
__device__ __forceinline__ void load_fragB(FragAB& b, const unsigned short* rowp, int hf) {
  const uint4* p = (const uint4*)(rowp + 8 * hf);
  uint4 lo = p[0], hi = p[2];
  b.d[0] = lo.x; b.d[1] = lo.y; b.d[2] = lo.z; b.d[3] = lo.w;
  b.d[4] = hi.x; b.d[5] = hi.y; b.d[6] = hi.z; b.d[7] = hi.w;
}

// ---- CDNA5 async global->LDS copy (GLOBAL_LOAD_ASYNC_TO_LDS_B128, ASYNCcnt) ----
__device__ __forceinline__ void async_copy_b128(void* lds, const void* gptr) {
  unsigned off = (unsigned)(size_t)lds;            // low 32 bits = LDS offset
  unsigned long long ga = (unsigned long long)(size_t)gptr;
  asm volatile("global_load_async_to_lds_b128 %0, %1, off"
               :: "v"(off), "v"(ga) : "memory");
}
__device__ __forceinline__ void wait_async() {
  asm volatile("s_wait_asynccnt 0x0" ::: "memory");
}

// ============================================================
// Kernel 1: projection GEMM  out_bf16[B,H,S,64] = (X @ W + b) * scale
// Tile 128(M) x 128(N) x 32(K); 256 threads = 8 waves; wave = 16x128 strip.
// ============================================================
__global__ __launch_bounds__(256)
void mha_proj_kernel(const float* __restrict__ X, const float* __restrict__ W,
                     const float* __restrict__ bias,
                     unsigned short* __restrict__ out, float scale) {
  __shared__ unsigned short As[128][LPAD];   // [m][k] bf16
  __shared__ unsigned short Bs[128][LPAD];   // [n][k] bf16 (transposed)

  const int t    = threadIdx.x;
  const int wave = t >> 5;
  const int lane = t & 31;
  const int lrow = lane & 15;
  const int hf   = lane >> 4;
  const int m0   = blockIdx.y * 128;
  const int n0   = blockIdx.x * 128;

  FragC c[8];
  const v8f zero = {0, 0, 0, 0, 0, 0, 0, 0};
#pragma unroll
  for (int nf = 0; nf < 8; ++nf) c[nf].v = zero;

  for (int k0 = 0; k0 < EMB; k0 += 32) {
    // stage X tile (128x32 f32 -> bf16)
#pragma unroll
    for (int i = 0; i < 4; ++i) {
      int e   = (t + i * 256) * 4;
      int row = e >> 5, col = e & 31;
      const float4 xv = *(const float4*)(X + (size_t)(m0 + row) * EMB + k0 + col);
      As[row][col + 0] = f2bf(xv.x); As[row][col + 1] = f2bf(xv.y);
      As[row][col + 2] = f2bf(xv.z); As[row][col + 3] = f2bf(xv.w);
    }
    // stage W tile (32x128 f32 -> bf16, transposed [n][k])
#pragma unroll
    for (int i = 0; i < 4; ++i) {
      int e = (t + i * 256) * 4;
      int k = e >> 7, n = e & 127;
      const float4 wv = *(const float4*)(W + (size_t)(k0 + k) * HDIM + n0 + n);
      Bs[n + 0][k] = f2bf(wv.x); Bs[n + 1][k] = f2bf(wv.y);
      Bs[n + 2][k] = f2bf(wv.z); Bs[n + 3][k] = f2bf(wv.w);
    }
    __syncthreads();

    FragAB a;
    load_fragA(a, &As[wave * 16 + lrow][0], hf);
#pragma unroll
    for (int nf = 0; nf < 8; ++nf) {
      FragAB b;
      load_fragB(b, &Bs[nf * 16 + lrow][0], hf);
      c[nf].v = wmma_bf16(a, b, c[nf].v);
    }
    __syncthreads();
  }

  // epilogue: +bias, *scale, bf16, head-split store [B,H,S,64]
#pragma unroll
  for (int nf = 0; nf < 8; ++nf) {
    int ng = n0 + nf * 16 + lrow;
    float bn = bias[ng];
    int h = ng >> 6, d = ng & 63;
#pragma unroll
    for (int r = 0; r < 8; ++r) {
      int mg = m0 + wave * 16 + r + 8 * hf;
      int bi = mg >> 11;             // / SEQ
      int s  = mg & (SEQ - 1);
      float val = (c[nf].f[r] + bn) * scale;
      out[((size_t)(bi * NH + h) * SEQ + s) * HD + d] = f2bf(val);
    }
  }
}

// ============================================================
// Kernel 2: causal flash attention.  Block = (b,h, q-tile of 64),
// 128 threads = 4 waves; wave owns 16 q rows; KV tiles of 32.
// ============================================================
__global__ __launch_bounds__(128)
void mha_attn_kernel(const unsigned short* __restrict__ qbuf,
                     const unsigned short* __restrict__ kbuf,
                     const unsigned short* __restrict__ vbuf,
                     unsigned short* __restrict__ abuf) {
  __shared__ unsigned short Ks[32][64];          // [kv][d], 128B rows
  __shared__ unsigned short Vt[64][LPAD];        // [d][kv] transposed
  __shared__ unsigned short Ps[4][16][LPAD];     // per-wave P tile

  const int t    = threadIdx.x;
  const int wave = t >> 5;
  const int lane = t & 31;
  const int lrow = lane & 15;
  const int hf   = lane >> 4;
  const int bh   = blockIdx.y;
  const int bi   = bh >> 4;
  const int h    = bh & 15;
  const int q0   = blockIdx.x * 64;
  const size_t base = (size_t)bh * SEQ * HD;

  // preload q A-fragments (head dim 64 = 2 x 32) straight from global (b128)
  FragAB qa[2];
  {
    const unsigned short* qp = qbuf + base + (size_t)(q0 + wave * 16 + lrow) * HD;
    load_fragA(qa[0], qp, hf);
    load_fragA(qa[1], qp + 32, hf);
  }

  float m_r[8], l_r[8];
  FragC o[4];
  const v8f zero = {0, 0, 0, 0, 0, 0, 0, 0};
#pragma unroll
  for (int r = 0; r < 8; ++r) { m_r[r] = -1e30f; l_r[r] = 0.f; }
#pragma unroll
  for (int nf = 0; nf < 4; ++nf) o[nf].v = zero;

  const int ntiles = (q0 >> 5) + 2;              // causal bound
  for (int tile = 0; tile < ntiles; ++tile) {
    const int k0 = tile * 32;
    // K tile: pure bf16 copy -> async DMA to LDS (ASYNCcnt path)
#pragma unroll
    for (int i = 0; i < 2; ++i) {
      int e  = (t + i * 128) * 8;
      int kv = e >> 6, d = e & 63;
      async_copy_b128(&Ks[kv][d], kbuf + base + (size_t)(k0 + kv) * HD + d);
    }
    // V tile transposed [d][kv]: needs shuffle, manual staging
#pragma unroll
    for (int i = 0; i < 2; ++i) {
      int e  = (t + i * 128) * 8;
      int kv = e >> 6, d = e & 63;
      Pack8 p;
      p.q = *(const uint4*)(vbuf + base + (size_t)(k0 + kv) * HD + d);
#pragma unroll
      for (int j = 0; j < 8; ++j) Vt[d + j][kv] = p.u[j];
    }
    wait_async();
    __syncthreads();

    // scores = q @ k^T  (16 x 32 per wave)
    FragC sc[2];
    sc[0].v = zero; sc[1].v = zero;
#pragma unroll
    for (int nf = 0; nf < 2; ++nf)
#pragma unroll
      for (int f = 0; f < 2; ++f) {
        FragAB b;
        load_fragB(b, &Ks[nf * 16 + lrow][f * 32], hf);
        sc[nf].v = wmma_bf16(qa[f], b, sc[nf].v);
      }

    // causal mask + online softmax (rows spread across 16-lane halves)
#pragma unroll
    for (int r = 0; r < 8; ++r) {
      int rowg = q0 + wave * 16 + r + 8 * hf;
      if (k0 + lrow > rowg)      sc[0].f[r] = -3e38f;
      if (k0 + 16 + lrow > rowg) sc[1].f[r] = -3e38f;
      float mt = fmaxf(sc[0].f[r], sc[1].f[r]);
      mt = fmaxf(mt, __shfl_xor(mt, 8, 32));
      mt = fmaxf(mt, __shfl_xor(mt, 4, 32));
      mt = fmaxf(mt, __shfl_xor(mt, 2, 32));
      mt = fmaxf(mt, __shfl_xor(mt, 1, 32));
      float mn = fmaxf(m_r[r], mt);
      float sc_old = __expf(m_r[r] - mn);
      m_r[r] = mn;
      float p0 = __expf(sc[0].f[r] - mn);
      float p1 = __expf(sc[1].f[r] - mn);
      float rs = p0 + p1;
      rs += __shfl_xor(rs, 8, 32);
      rs += __shfl_xor(rs, 4, 32);
      rs += __shfl_xor(rs, 2, 32);
      rs += __shfl_xor(rs, 1, 32);
      l_r[r] = l_r[r] * sc_old + rs;
      Ps[wave][r + 8 * hf][lrow]      = f2bf(p0);
      Ps[wave][r + 8 * hf][16 + lrow] = f2bf(p1);
#pragma unroll
      for (int nf = 0; nf < 4; ++nf) o[nf].f[r] *= sc_old;
    }

    // O += P @ V  (re-read P from LDS in A-layout, b128)
    FragAB pa;
    load_fragA(pa, &Ps[wave][lrow][0], hf);
#pragma unroll
    for (int nf = 0; nf < 4; ++nf) {
      FragAB b;
      load_fragB(b, &Vt[nf * 16 + lrow][0], hf);
      o[nf].v = wmma_bf16(pa, b, o[nf].v);
    }
    __syncthreads();
  }

  // epilogue: normalize, store bf16 to [B,S,H*D]
#pragma unroll
  for (int nf = 0; nf < 4; ++nf) {
    int d = nf * 16 + lrow;
#pragma unroll
    for (int r = 0; r < 8; ++r) {
      int sg = q0 + wave * 16 + r + 8 * hf;
      float val = o[nf].f[r] / l_r[r];
      abuf[(size_t)(bi * SEQ + sg) * HDIM + h * HD + d] = f2bf(val);
    }
  }
}

// ============================================================
// Kernel 3: output projection  out_f32 = attn_bf16 @ W_o + b_o
// Tile 128 x 128 x 32; A tile staged via async DMA (already bf16).
// ============================================================
__global__ __launch_bounds__(256)
void mha_outproj_kernel(const unsigned short* __restrict__ Abf,
                        const float* __restrict__ W,
                        const float* __restrict__ bias,
                        float* __restrict__ out) {
  __shared__ unsigned short As[128][LPAD];
  __shared__ unsigned short Bs[128][LPAD];

  const int t    = threadIdx.x;
  const int wave = t >> 5;
  const int lane = t & 31;
  const int lrow = lane & 15;
  const int hf   = lane >> 4;
  const int m0   = blockIdx.y * 128;
  const int n0   = blockIdx.x * 128;

  FragC c[8];
  const v8f zero = {0, 0, 0, 0, 0, 0, 0, 0};
#pragma unroll
  for (int nf = 0; nf < 8; ++nf) c[nf].v = zero;

  for (int k0 = 0; k0 < HDIM; k0 += 32) {
    // stage A tile (already bf16): async DMA, b128 per lane
#pragma unroll
    for (int i = 0; i < 2; ++i) {
      int e   = (t + i * 256) * 8;
      int row = e >> 5, col = e & 31;
      async_copy_b128(&As[row][col],
                      Abf + (size_t)(m0 + row) * HDIM + k0 + col);
    }
    // stage W_o tile (32x128 f32 -> bf16, transposed)
#pragma unroll
    for (int i = 0; i < 4; ++i) {
      int e = (t + i * 256) * 4;
      int k = e >> 7, n = e & 127;
      const float4 wv = *(const float4*)(W + (size_t)(k0 + k) * EMB + n0 + n);
      Bs[n + 0][k] = f2bf(wv.x); Bs[n + 1][k] = f2bf(wv.y);
      Bs[n + 2][k] = f2bf(wv.z); Bs[n + 3][k] = f2bf(wv.w);
    }
    wait_async();
    __syncthreads();

    FragAB a;
    load_fragA(a, &As[wave * 16 + lrow][0], hf);
#pragma unroll
    for (int nf = 0; nf < 8; ++nf) {
      FragAB b;
      load_fragB(b, &Bs[nf * 16 + lrow][0], hf);
      c[nf].v = wmma_bf16(a, b, c[nf].v);
    }
    __syncthreads();
  }

#pragma unroll
  for (int nf = 0; nf < 8; ++nf) {
    int ng = n0 + nf * 16 + lrow;
    float bn = bias[ng];
#pragma unroll
    for (int r = 0; r < 8; ++r) {
      int mg = m0 + wave * 16 + r + 8 * hf;
      out[(size_t)mg * EMB + ng] = c[nf].f[r] + bn;
    }
  }
}

// ============================================================
extern "C" void kernel_launch(void* const* d_in, const int* in_sizes, int n_in,
                              void* d_out, int out_size, void* d_ws, size_t ws_size,
                              hipStream_t stream) {
  (void)in_sizes; (void)n_in; (void)out_size; (void)ws_size;
  const float* Q  = (const float*)d_in[0];
  const float* K  = (const float*)d_in[1];
  const float* V  = (const float*)d_in[2];
  const float* Wq = (const float*)d_in[3];
  const float* bq = (const float*)d_in[4];
  const float* Wk = (const float*)d_in[5];
  const float* bk = (const float*)d_in[6];
  const float* Wv = (const float*)d_in[7];
  const float* bv = (const float*)d_in[8];
  const float* Wo = (const float*)d_in[9];
  const float* bo = (const float*)d_in[10];
  float* out = (float*)d_out;

  // workspace: 4 bf16 buffers of B*S*HDIM elements (64 MB total)
  const size_t elems = (size_t)BATCH * SEQ * HDIM;
  unsigned short* qb = (unsigned short*)d_ws;
  unsigned short* kb = qb + elems;
  unsigned short* vb = kb + elems;
  unsigned short* ab = vb + elems;

  dim3 gProj(HDIM / 128, (BATCH * SEQ) / 128);   // (8, 64)
  mha_proj_kernel<<<gProj, 256, 0, stream>>>(Q, Wq, bq, qb, 0.125f); // fold 1/sqrt(64)
  mha_proj_kernel<<<gProj, 256, 0, stream>>>(K, Wk, bk, kb, 1.0f);
  mha_proj_kernel<<<gProj, 256, 0, stream>>>(V, Wv, bv, vb, 1.0f);

  dim3 gAttn(SEQ / 64, BATCH * NH);              // (32, 64)
  mha_attn_kernel<<<gAttn, 128, 0, stream>>>(qb, kb, vb, ab);

  dim3 gOut(EMB / 128, (BATCH * SEQ) / 128);
  mha_outproj_kernel<<<gOut, 256, 0, stream>>>(ab, Wo, bo, out);
}
